// PillarMVFFeatureNet_78537771975379
// MI455X (gfx1250) — matrix operand order repaired and checked by reading, not compiled, not run on hardware
//
#include <hip/hip_runtime.h>

// ---------------------------------------------------------------------------
// PillarMVFFeatureNet for MI455X (gfx1250): bf16 WMMA (16x16x32) everywhere.
// Round 2: pre-swizzled bf16 weights (direct global WMMA-B fragments),
// vectorized b128 fragment/staging paths, LDS-cached BN scale/bias, prefetch.
// ---------------------------------------------------------------------------

typedef __bf16 bf16;
typedef __attribute__((ext_vector_type(16))) __bf16 v16bf;
typedef __attribute__((ext_vector_type(8)))  __bf16 v8bf;
typedef __attribute__((ext_vector_type(8)))  float  v8f;

#define NPTS 160000
#define BB   2
#define GNX  352
#define GNY  400
#define NSEG (BB * GNY * GNX)   // 281600
#define F_XMIN 0.0f
#define F_YMIN (-40.0f)
#define F_ZMIN (-3.0f)
#define F_XMAX 70.4f
#define F_YMAX 40.0f
#define F_VX 0.2f
#define F_VY 0.2f
#define F_VZ 4.0f
#define F_EPS 1e-3f

#define WMMA_BF16(a, b, c) \
  __builtin_amdgcn_wmma_f32_16x16x32_bf16(false, (a), false, (b), (short)0, (c), false, false)

__device__ inline v8f zero8() {
  v8f z = {0.f, 0.f, 0.f, 0.f, 0.f, 0.f, 0.f, 0.f};
  return z;
}
__device__ inline v8bf zerobf8() {
  v8bf z;
#pragma unroll
  for (int i = 0; i < 8; ++i) z[i] = (bf16)0.f;
  return z;
}

// A fragment: 16x32 bf16 tile, row m at base + m*lda. Per ISA layout each lane
// needs rows' k = [8h..8h+7] and [16+8h..23+8h]: two contiguous 16B runs.
__device__ inline v16bf frag_a(const bf16* base, int lda, int lane) {
  int m = lane & 15, h = lane >> 4;
  const bf16* row = base + m * lda + 8 * h;
  v8bf lo = *(const v8bf*)(row);
  v8bf hi = *(const v8bf*)(row + 16);
  return __builtin_shufflevector(lo, hi, 0, 1, 2, 3, 4, 5, 6, 7,
                                 8, 9, 10, 11, 12, 13, 14, 15);
}

// B fragment from a pre-swizzled region: region holds one (32K x 16N) tile as
// [lane][16] contiguous (lane = 16*((k>>3)&1) + n, slot = (k&7) + 8*((k>>4)&1)) -> 32B per lane.
__device__ inline v16bf frag_b_g(const bf16* region, int lane) {
  const bf16* p = region + lane * 16;
  v8bf lo = *(const v8bf*)(p);
  v8bf hi = *(const v8bf*)(p + 8);
  return __builtin_shufflevector(lo, hi, 0, 1, 2, 3, 4, 5, 6, 7,
                                 8, 9, 10, 11, 12, 13, 14, 15);
}

// swizzled destination index for element (k, n) of a Kx128 weight slab
__device__ inline int swz_idx(int k, int n) {
  int kc = k >> 5, t = n >> 4;
  int lane = ((k >> 3) & 1) * 16 + (n & 15);
  int slot = (k & 7) + (((k >> 4) & 1) << 3);
  return ((kc * 8 + t) * 32 + lane) * 16 + slot;
}

// atomic max for non-negative floats (bit pattern compare == float compare)
__device__ inline void atomicMaxPosF(float* p, float v) {
  atomicMax((int*)p, __float_as_int(v));
}

// ---------------------------------------------------------------------------
// K0: weight prep: f32 -> bf16, pre-swizzled into WMMA-B fragment order.
// mode 0: GEMM W (Ktot x 128 row-major), zero-padded to Kpad (total=Kpad*128)
// mode 1: conv3 OIHW 128x128x3x3 -> 9 slabs of [128K x 128N]
// mode 2: wd OxI -> one slab (B[k][n] = w[n*128+k])
// mode 3: deconv CxOxKKxKK (KK in xdim) -> KK*KK slabs
// mode 4: fconv 128x384x3x3 -> 27 slabs ordered [src][ky][kx]
// ---------------------------------------------------------------------------
__global__ __launch_bounds__(256) void k_prep(const float* __restrict__ src,
                                              bf16* __restrict__ dst,
                                              int mode, int total, int xdim) {
  int idx = blockIdx.x * 256 + threadIdx.x;
  if (idx >= total) return;
  float val = 0.f;
  int slab = 0, k = 0, n = 0;
  if (mode == 0) {
    k = idx >> 7; n = idx & 127;
    val = (k < xdim) ? src[k * 128 + n] : 0.f;
  } else if (mode == 1) {
    slab = idx >> 14; int r = idx & 16383;
    k = r >> 7; n = r & 127;
    int ky = slab / 3, kx = slab % 3;
    val = src[((n * 128 + k) * 3 + ky) * 3 + kx];
  } else if (mode == 2) {
    k = idx >> 7; n = idx & 127;
    val = src[n * 128 + k];
  } else if (mode == 3) {
    slab = idx >> 14; int r = idx & 16383;
    k = r >> 7; n = r & 127;
    int si = slab / xdim, sj = slab % xdim;
    val = src[((k * 128 + n) * xdim + si) * xdim + sj];
  } else {  // mode 4
    slab = idx >> 14; int r = idx & 16383;
    k = r >> 7; n = r & 127;
    int s3 = slab / 9, pos = slab % 9;
    int ky = pos / 3, kx = pos % 3;
    val = src[((n * 384 + s3 * 128 + k) * 3 + ky) * 3 + kx];
  }
  dst[slab * 16384 + swz_idx(k, n)] = (bf16)val;
}

// ---------------------------------------------------------------------------
// K1: per-point scatter of count / sum(xyz) / sum(pairs) via f32 atomics
// ---------------------------------------------------------------------------
__global__ __launch_bounds__(256) void k_scatter(const float* __restrict__ feat,
                                                 const int* __restrict__ coors,
                                                 float* __restrict__ cnt,
                                                 float* __restrict__ sums) {
  int p = blockIdx.x * 256 + threadIdx.x;
  if (p >= NPTS) return;
  float x = feat[4 * p], y = feat[4 * p + 1], z = feat[4 * p + 2];
  int bi = coors[4 * p], yi = coors[4 * p + 2], xi = coors[4 * p + 3];
  int seg = (bi * GNY + yi) * GNX + xi;
  atomicAdd(&cnt[seg], 1.f);
  atomicAdd(&sums[0 * NSEG + seg], x);
  atomicAdd(&sums[1 * NSEG + seg], y);
  atomicAdd(&sums[2 * NSEG + seg], z);
  atomicAdd(&sums[3 * NSEG + seg], x * x);
  atomicAdd(&sums[4 * NSEG + seg], x * y);
  atomicAdd(&sums[5 * NSEG + seg], x * z);
  atomicAdd(&sums[6 * NSEG + seg], y * y);
  atomicAdd(&sums[7 * NSEG + seg], y * z);
  atomicAdd(&sums[8 * NSEG + seg], z * z);
}

// ---------------------------------------------------------------------------
// K2: per-point 20-dim stats -> GEMM (20->128 padded to 32) -> bn -> relu -> h
// ---------------------------------------------------------------------------
__global__ __launch_bounds__(256) void k_pn1(const float* __restrict__ feat,
                                             const int* __restrict__ coors,
                                             const float* __restrict__ cnt,
                                             const float* __restrict__ sums,
                                             const bf16* __restrict__ wz,  // swizzled 32x128
                                             const float* __restrict__ g, const float* __restrict__ bt,
                                             const float* __restrict__ mu, const float* __restrict__ vr,
                                             bf16* __restrict__ hout) {
  __shared__ __align__(16) bf16 sA[128 * 32];
  __shared__ float sS[128], sO[128];
  int tid = threadIdx.x;
  if (tid < 128) {
    int p = blockIdx.x * 128 + tid;
    float fx = feat[4 * p], fy = feat[4 * p + 1], fz = feat[4 * p + 2], fi = feat[4 * p + 3];
    int bi = coors[4 * p], zi = coors[4 * p + 1], yi = coors[4 * p + 2], xi = coors[4 * p + 3];
    int seg = (bi * GNY + yi) * GNX + xi;
    float c = cnt[seg];
    float inv = 1.f / fmaxf(c, 1.f);
    float mx = sums[0 * NSEG + seg] * inv;
    float my = sums[1 * NSEG + seg] * inv;
    float mz = sums[2 * NSEG + seg] * inv;
    float vcx = (xi + 0.5f) * F_VX + F_XMIN;
    float vcy = (yi + 0.5f) * F_VY + F_YMIN;
    float vcz = (zi + 0.5f) * F_VZ + F_ZMIN;
    float st[20];
    st[0] = mx; st[1] = my; st[2] = mz;
    st[3] = fx - mx; st[4] = fy - my; st[5] = fz - mz;
    st[6]  = sums[3 * NSEG + seg] * inv - mx * mx;
    st[7]  = sums[4 * NSEG + seg] * inv - mx * my;
    st[8]  = sums[5 * NSEG + seg] * inv - mx * mz;
    st[9]  = sums[6 * NSEG + seg] * inv - my * my;
    st[10] = sums[7 * NSEG + seg] * inv - my * mz;
    st[11] = sums[8 * NSEG + seg] * inv - mz * mz;
    st[12] = vcx; st[13] = vcy; st[14] = vcz;
    st[15] = c;
    st[16] = fx - vcx; st[17] = fy - vcy; st[18] = fz - vcz;
    st[19] = fi;
    bf16* row = &sA[tid * 32];
#pragma unroll
    for (int k = 0; k < 20; ++k) row[k] = (bf16)st[k];
#pragma unroll
    for (int k = 20; k < 32; ++k) row[k] = (bf16)0.f;
    float s = g[tid] * rsqrtf(vr[tid] + F_EPS);
    sS[tid] = s;
    sO[tid] = bt[tid] - mu[tid] * s;
  }
  __syncthreads();

  int wv = tid >> 5, lane = tid & 31;
  int n15 = lane & 15, hlf = lane >> 4;
  v16bf a = frag_a(&sA[wv * 16 * 32], 32, lane);
#pragma unroll
  for (int nt = 0; nt < 8; ++nt) {
    v8f acc = zero8();
    acc = WMMA_BF16(a, frag_b_g(wz + nt * 512, lane), acc);
    int n = nt * 16 + n15;
    float s = sS[n], bia = sO[n];
#pragma unroll
    for (int r = 0; r < 8; ++r) {
      long row = (long)blockIdx.x * 128 + wv * 16 + r + 8 * hlf;
      hout[row * 128 + n] = (bf16)fmaxf(acc[r] * s + bia, 0.f);
    }
  }
}

// ---------------------------------------------------------------------------
// K3/K9: rows x (K=128) GEMM with swizzled global B.
//   mode 0: store bf16; mode 1: atomicMax scatter into out_max[seg*128+n]
// ---------------------------------------------------------------------------
__global__ __launch_bounds__(256) void k_gemm128(const bf16* __restrict__ A,
                                                 const bf16* __restrict__ wz,  // swizzled 128x128
                                                 const float* __restrict__ g, const float* __restrict__ bt,
                                                 const float* __restrict__ mu, const float* __restrict__ vr,
                                                 int mode, bf16* __restrict__ out_bf,
                                                 float* __restrict__ out_max,
                                                 const int* __restrict__ coors) {
  __shared__ __align__(16) bf16 sA[128 * 128];
  __shared__ float sS[128], sO[128];
  int tid = threadIdx.x;
  long rowbase = (long)blockIdx.x * 128;
  for (int idx = tid; idx < 128 * 16; idx += 256) {
    int row = idx >> 4, cc = (idx & 15) * 8;
    *(v8bf*)&sA[row * 128 + cc] = *(const v8bf*)&A[(rowbase + row) * 128 + cc];
  }
  if (tid < 128) {
    float s = g[tid] * rsqrtf(vr[tid] + F_EPS);
    sS[tid] = s;
    sO[tid] = bt[tid] - mu[tid] * s;
  }
  __syncthreads();

  int wv = tid >> 5, lane = tid & 31;
  v8f acc[8];
#pragma unroll
  for (int t = 0; t < 8; ++t) acc[t] = zero8();
#pragma unroll
  for (int kc = 0; kc < 4; ++kc) {
    v16bf a = frag_a(&sA[(wv * 16) * 128 + kc * 32], 128, lane);
#pragma unroll
    for (int nt = 0; nt < 8; ++nt)
      acc[nt] = WMMA_BF16(a, frag_b_g(wz + (kc * 8 + nt) * 512, lane), acc[nt]);
  }
  int n15 = lane & 15, hlf = lane >> 4;
#pragma unroll
  for (int nt = 0; nt < 8; ++nt) {
    int n = nt * 16 + n15;
    float s = sS[n], bia = sO[n];
#pragma unroll
    for (int r = 0; r < 8; ++r) {
      long row = rowbase + wv * 16 + r + 8 * hlf;
      float val = fmaxf(acc[nt][r] * s + bia, 0.f);
      if (mode == 0) {
        out_bf[row * 128 + n] = (bf16)val;
      } else {
        int bi = coors[4 * row], yi = coors[4 * row + 2], xi = coors[4 * row + 3];
        int seg = (bi * GNY + yi) * GNX + xi;
        atomicMaxPosF(&out_max[(long)seg * 128 + n], val);
      }
    }
  }
}

// ---------------------------------------------------------------------------
// K4: f32 canvas -> bf16 canvas
// ---------------------------------------------------------------------------
__global__ __launch_bounds__(256) void k_cvt(const float* __restrict__ src,
                                             bf16* __restrict__ dst) {
  long i0 = ((long)blockIdx.x * 256 + threadIdx.x) * 8;
  v8bf o;
#pragma unroll
  for (int j = 0; j < 8; ++j) o[j] = (bf16)src[i0 + j];
  *(v8bf*)&dst[i0] = o;
}

// ---------------------------------------------------------------------------
// K5: 3x3 conv (NHWC, 128->128) implicit GEMM; swizzled global B fragments.
// STRIDE=1 -> TM=32, STRIDE=2 -> TM=16 (shared 3x34x128 activation stage).
// ---------------------------------------------------------------------------
template <int STRIDE, int TM>
__global__ __launch_bounds__(256) void k_conv3(const bf16* __restrict__ in, int Hin, int Win,
                                               bf16* __restrict__ out, int Hout, int Wout,
                                               const bf16* __restrict__ wz,  // 9 swizzled slabs
                                               const float* __restrict__ g, const float* __restrict__ bt,
                                               const float* __restrict__ mu, const float* __restrict__ vr,
                                               const bf16* __restrict__ idn, int relu_flag) {
  __shared__ __align__(16) bf16 sAct[3 * 34 * 128];
  __shared__ float sS[128], sO[128];
  int tid = threadIdx.x;
  int strips = (Wout + TM - 1) / TM;
  int bIdx = blockIdx.x / (Hout * strips);
  int rem  = blockIdx.x % (Hout * strips);
  int y = rem / strips;
  int x0 = (rem % strips) * TM;
  int y0 = y * STRIDE - 1;
  int xin0 = x0 * STRIDE - 1;

  for (int idx = tid; idx < 3 * 34 * 16; idx += 256) {
    int r = idx / (34 * 16);
    int rm = idx % (34 * 16);
    int xx = rm >> 4, cc = (rm & 15) * 8;
    int yy = y0 + r, xg = xin0 + xx;
    v8bf v = zerobf8();
    if (yy >= 0 && yy < Hin && xg >= 0 && xg < Win)
      v = *(const v8bf*)&in[(((long)bIdx * Hin + yy) * Win + xg) * 128 + cc];
    *(v8bf*)&sAct[(r * 34 + xx) * 128 + cc] = v;
  }
  if (tid < 128) {
    float s = g[tid] * rsqrtf(vr[tid] + F_EPS);
    sS[tid] = s;
    sO[tid] = bt[tid] - mu[tid] * s;
  }
  __syncthreads();

  int wv = tid >> 5, lane = tid & 31;
  constexpr int NACC = (TM == 32) ? 2 : 1;
  int rt  = (TM == 32) ? (wv & 1) : 0;
  int nt0 = (TM == 32) ? ((wv >> 1) * 2) : wv;
  v8f acc[NACC];
#pragma unroll
  for (int t = 0; t < NACC; ++t) acc[t] = zero8();

  for (int pos = 0; pos < 9; ++pos) {
    int ky = pos / 3, kx = pos % 3;
    __builtin_prefetch(wz + ((pos + 1) % 9) * 16384 + tid * 64, 0, 1);
    const bf16* wslab = wz + pos * 16384;
#pragma unroll
    for (int kc = 0; kc < 4; ++kc) {
      const bf16* abase = &sAct[(ky * 34 + rt * 16 * STRIDE + kx) * 128 + kc * 32];
      v16bf a = frag_a(abase, STRIDE * 128, lane);
#pragma unroll
      for (int t = 0; t < NACC; ++t)
        acc[t] = WMMA_BF16(a, frag_b_g(wslab + (kc * 8 + nt0 + t) * 512, lane), acc[t]);
    }
  }

  int n15 = lane & 15, hlf = lane >> 4;
#pragma unroll
  for (int t = 0; t < NACC; ++t) {
    int n = (nt0 + t) * 16 + n15;
    float s = sS[n], bia = sO[n];
#pragma unroll
    for (int r = 0; r < 8; ++r) {
      int px = x0 + rt * 16 + r + 8 * hlf;
      if (px < Wout) {
        long oidx = (((long)bIdx * Hout + y) * Wout + px) * 128 + n;
        float val = acc[t][r] * s + bia;
        if (idn) val += (float)idn[oidx];
        if (relu_flag) val = fmaxf(val, 0.f);
        out[oidx] = (bf16)val;
      }
    }
  }
}

// ---------------------------------------------------------------------------
// K6: 1x1 stride-2 conv (downsample path), bn only, no relu.
// ---------------------------------------------------------------------------
__global__ __launch_bounds__(256) void k_wd(const bf16* __restrict__ in, int Hin, int Win,
                                            bf16* __restrict__ out, int Hout, int Wout,
                                            const bf16* __restrict__ wz,  // swizzled 128x128
                                            const float* __restrict__ g, const float* __restrict__ bt,
                                            const float* __restrict__ mu, const float* __restrict__ vr) {
  __shared__ __align__(16) bf16 sA[32 * 128];
  __shared__ float sS[128], sO[128];
  int tid = threadIdx.x;
  int strips = (Wout + 31) / 32;
  int bIdx = blockIdx.x / (Hout * strips);
  int rem  = blockIdx.x % (Hout * strips);
  int y = rem / strips;
  int x0 = (rem % strips) * 32;

  for (int idx = tid; idx < 32 * 16; idx += 256) {
    int m = idx >> 4, cc = (idx & 15) * 8;
    int xg = 2 * (x0 + m), yg = 2 * y;
    v8bf v = zerobf8();
    if (xg < Win && yg < Hin) v = *(const v8bf*)&in[(((long)bIdx * Hin + yg) * Win + xg) * 128 + cc];
    *(v8bf*)&sA[m * 128 + cc] = v;
  }
  if (tid < 128) {
    float s = g[tid] * rsqrtf(vr[tid] + F_EPS);
    sS[tid] = s;
    sO[tid] = bt[tid] - mu[tid] * s;
  }
  __syncthreads();

  int wv = tid >> 5, lane = tid & 31;
  int rt = wv & 1, nt0 = (wv >> 1) * 2;
  v8f acc[2] = {zero8(), zero8()};
#pragma unroll
  for (int kc = 0; kc < 4; ++kc) {
    v16bf a = frag_a(&sA[rt * 16 * 128 + kc * 32], 128, lane);
#pragma unroll
    for (int t = 0; t < 2; ++t)
      acc[t] = WMMA_BF16(a, frag_b_g(wz + (kc * 8 + nt0 + t) * 512, lane), acc[t]);
  }
  int n15 = lane & 15, hlf = lane >> 4;
#pragma unroll
  for (int t = 0; t < 2; ++t) {
    int n = (nt0 + t) * 16 + n15;
    float s = sS[n], bia = sO[n];
#pragma unroll
    for (int r = 0; r < 8; ++r) {
      int px = x0 + rt * 16 + r + 8 * hlf;
      if (px < Wout)
        out[(((long)bIdx * Hout + y) * Wout + px) * 128 + n] = (bf16)(acc[t][r] * s + bia);
    }
  }
}

// ---------------------------------------------------------------------------
// K7: deconv: up[b, KK*y+i, KK*x+j, o] = sum_c in[b,y,x,c] * W[c,o,i,j]
// ---------------------------------------------------------------------------
template <int KK>
__global__ __launch_bounds__(256) void k_deconv(const bf16* __restrict__ in, int Hin, int Win,
                                                bf16* __restrict__ out, int Hout, int Wout,
                                                const bf16* __restrict__ wz) {  // KK*KK slabs
  __shared__ __align__(16) bf16 sA[32 * 128];
  int tid = threadIdx.x;
  int strips = (Win + 31) / 32;
  int bIdx = blockIdx.x / (Hin * strips);
  int rem  = blockIdx.x % (Hin * strips);
  int y = rem / strips;
  int x0 = (rem % strips) * 32;

  for (int idx = tid; idx < 32 * 16; idx += 256) {
    int m = idx >> 4, cc = (idx & 15) * 8;
    int xg = x0 + m;
    v8bf v = zerobf8();
    if (xg < Win) v = *(const v8bf*)&in[(((long)bIdx * Hin + y) * Win + xg) * 128 + cc];
    *(v8bf*)&sA[m * 128 + cc] = v;
  }
  __syncthreads();

  int wv = tid >> 5, lane = tid & 31;
  int rt = wv & 1, nt0 = (wv >> 1) * 2;
  int n15 = lane & 15, hlf = lane >> 4;

  for (int sub = 0; sub < KK * KK; ++sub) {
    int si = sub / KK, sj = sub % KK;
    __builtin_prefetch(wz + ((sub + 1) % (KK * KK)) * 16384 + tid * 64, 0, 1);
    const bf16* wslab = wz + sub * 16384;
    v8f acc[2] = {zero8(), zero8()};
#pragma unroll
    for (int kc = 0; kc < 4; ++kc) {
      v16bf a = frag_a(&sA[rt * 16 * 128 + kc * 32], 128, lane);
#pragma unroll
      for (int t = 0; t < 2; ++t)
        acc[t] = WMMA_BF16(a, frag_b_g(wslab + (kc * 8 + nt0 + t) * 512, lane), acc[t]);
    }
#pragma unroll
    for (int t = 0; t < 2; ++t) {
      int n = (nt0 + t) * 16 + n15;
#pragma unroll
      for (int r = 0; r < 8; ++r) {
        int px = x0 + rt * 16 + r + 8 * hlf;
        if (px < Win)
          out[(((long)bIdx * Hout + KK * y + si) * Wout + KK * px + sj) * 128 + n] = (bf16)acc[t][r];
      }
    }
  }
}

// ---------------------------------------------------------------------------
// K8: fusion conv: 3 sources x 128ch, 3x3, + bias, no activation.
// ---------------------------------------------------------------------------
__global__ __launch_bounds__(256) void k_fconv(const bf16* __restrict__ in0,
                                               const bf16* __restrict__ in1,
                                               const bf16* __restrict__ in2,
                                               const bf16* __restrict__ wz,  // 27 slabs [src][ky][kx]
                                               const float* __restrict__ bias,
                                               bf16* __restrict__ out) {
  __shared__ __align__(16) bf16 sAct[3 * 34 * 128];
  int tid = threadIdx.x;
  const int strips = (GNX + 31) / 32;  // 11
  int bIdx = blockIdx.x / (GNY * strips);
  int rem  = blockIdx.x % (GNY * strips);
  int y = rem / strips;
  int x0 = (rem % strips) * 32;
  int y0 = y - 1, xin0 = x0 - 1;

  int wv = tid >> 5, lane = tid & 31;
  int rt = wv & 1, nt0 = (wv >> 1) * 2;
  v8f acc[2] = {zero8(), zero8()};

  for (int src = 0; src < 3; ++src) {
    const bf16* in = (src == 0) ? in0 : ((src == 1) ? in1 : in2);
    __syncthreads();
    for (int idx = tid; idx < 3 * 34 * 16; idx += 256) {
      int r = idx / (34 * 16);
      int rm = idx % (34 * 16);
      int xx = rm >> 4, cc = (rm & 15) * 8;
      int yy = y0 + r, xg = xin0 + xx;
      v8bf v = zerobf8();
      if (yy >= 0 && yy < GNY && xg >= 0 && xg < GNX)
        v = *(const v8bf*)&in[(((long)bIdx * GNY + yy) * GNX + xg) * 128 + cc];
      *(v8bf*)&sAct[(r * 34 + xx) * 128 + cc] = v;
    }
    __syncthreads();
    for (int pos = 0; pos < 9; ++pos) {
      int ky = pos / 3, kx = pos % 3;
      __builtin_prefetch(wz + ((src * 9 + pos + 1) % 27) * 16384 + tid * 64, 0, 1);
      const bf16* wslab = wz + (src * 9 + pos) * 16384;
#pragma unroll
      for (int kc = 0; kc < 4; ++kc) {
        const bf16* abase = &sAct[(ky * 34 + rt * 16 + kx) * 128 + kc * 32];
        v16bf a = frag_a(abase, 128, lane);
#pragma unroll
        for (int t = 0; t < 2; ++t)
          acc[t] = WMMA_BF16(a, frag_b_g(wslab + (kc * 8 + nt0 + t) * 512, lane), acc[t]);
      }
    }
  }
  int n15 = lane & 15, hlf = lane >> 4;
#pragma unroll
  for (int t = 0; t < 2; ++t) {
    int n = (nt0 + t) * 16 + n15;
    float bia = bias[n];
#pragma unroll
    for (int r = 0; r < 8; ++r) {
      int px = x0 + rt * 16 + r + 8 * hlf;
      if (px < GNX)
        out[(((long)bIdx * GNY + y) * GNX + px) * 128 + n] = (bf16)(acc[t][r] + bia);
    }
  }
}

// ---------------------------------------------------------------------------
// K10: per-point bilinear gather from fused map -> pts_view (bf16)
// ---------------------------------------------------------------------------
__global__ __launch_bounds__(256) void k_gather(const float* __restrict__ feat,
                                                const int* __restrict__ coors,
                                                const bf16* __restrict__ o,
                                                bf16* __restrict__ pv) {
  int t = blockIdx.x * 256 + threadIdx.x;
  if (t >= NPTS * 16) return;
  int p = t >> 4;
  int c0 = (t & 15) * 8;
  float x_ = feat[4 * p], y_ = feat[4 * p + 1];
  int bi = coors[4 * p];
  float gx = (x_ - F_XMIN) / ((F_XMAX - F_XMIN) * 0.5f) - 1.f;
  float gy = (y_ - F_YMIN) / ((F_YMAX - F_YMIN) * 0.5f) - 1.f;
  float ix = ((gx + 1.f) * GNX - 1.f) * 0.5f;
  float iy = ((gy + 1.f) * GNY - 1.f) * 0.5f;
  float x0f = floorf(ix), y0f = floorf(iy);
  float wx = ix - x0f, wy = iy - y0f;
  int x0 = (int)x0f, y0 = (int)y0f;
  float wq[4] = {(1.f - wx) * (1.f - wy), wx * (1.f - wy), (1.f - wx) * wy, wx * wy};
  int xs[4] = {x0, x0 + 1, x0, x0 + 1};
  int ys[4] = {y0, y0, y0 + 1, y0 + 1};
  float acc[8] = {0.f, 0.f, 0.f, 0.f, 0.f, 0.f, 0.f, 0.f};
#pragma unroll
  for (int q = 0; q < 4; ++q) {
    if (xs[q] >= 0 && xs[q] < GNX && ys[q] >= 0 && ys[q] < GNY) {
      v8bf vv = *(const v8bf*)&o[(((long)bi * GNY + ys[q]) * GNX + xs[q]) * 128 + c0];
      float wqq = wq[q];
#pragma unroll
      for (int j = 0; j < 8; ++j) acc[j] += wqq * (float)vv[j];
    }
  }
  v8bf ov;
#pragma unroll
  for (int j = 0; j < 8; ++j) ov[j] = (bf16)acc[j];
  *(v8bf*)&pv[(long)p * 128 + c0] = ov;
}

// ---------------------------------------------------------------------------
// K11: fm = relu(bn(concat[pts_view, h2] @ pn3_w)); segment-max into d_out
// ---------------------------------------------------------------------------
__global__ __launch_bounds__(256) void k_pn3(const bf16* __restrict__ Apv,
                                             const bf16* __restrict__ Ah2,
                                             const bf16* __restrict__ wz,  // swizzled 256x128
                                             const float* __restrict__ g, const float* __restrict__ bt,
                                             const float* __restrict__ mu, const float* __restrict__ vr,
                                             const int* __restrict__ coors,
                                             float* __restrict__ outmax) {
  __shared__ __align__(16) bf16 sA[128 * 128];
  __shared__ float sS[128], sO[128];
  int tid = threadIdx.x;
  long rowbase = (long)blockIdx.x * 128;
  int wv = tid >> 5, lane = tid & 31;
  if (tid < 128) {
    float s = g[tid] * rsqrtf(vr[tid] + F_EPS);
    sS[tid] = s;
    sO[tid] = bt[tid] - mu[tid] * s;
  }
  v8f acc[8];
#pragma unroll
  for (int t = 0; t < 8; ++t) acc[t] = zero8();

  for (int kh = 0; kh < 2; ++kh) {
    const bf16* src = kh ? Ah2 : Apv;
    __syncthreads();
    for (int idx = tid; idx < 128 * 16; idx += 256) {
      int row = idx >> 4, cc = (idx & 15) * 8;
      *(v8bf*)&sA[row * 128 + cc] = *(const v8bf*)&src[(rowbase + row) * 128 + cc];
    }
    __syncthreads();
#pragma unroll
    for (int kc = 0; kc < 4; ++kc) {
      v16bf a = frag_a(&sA[(wv * 16) * 128 + kc * 32], 128, lane);
#pragma unroll
      for (int nt = 0; nt < 8; ++nt)
        acc[nt] = WMMA_BF16(a, frag_b_g(wz + ((kh * 4 + kc) * 8 + nt) * 512, lane), acc[nt]);
    }
  }
  int n15 = lane & 15, hlf = lane >> 4;
#pragma unroll
  for (int nt = 0; nt < 8; ++nt) {
    int n = nt * 16 + n15;
    float s = sS[n], bia = sO[n];
#pragma unroll
    for (int r = 0; r < 8; ++r) {
      long row = rowbase + wv * 16 + r + 8 * hlf;
      float val = fmaxf(acc[nt][r] * s + bia, 0.f);
      int bi = coors[4 * row], yi = coors[4 * row + 2], xi = coors[4 * row + 3];
      int seg = (bi * GNY + yi) * GNX + xi;
      atomicMaxPosF(&outmax[(long)seg * 128 + n], val);
    }
  }
}

// ---------------------------------------------------------------------------
// Host orchestration
// ---------------------------------------------------------------------------
// Input leaf order (setup_inputs dict insertion order, tuples expanded):
//  0 features, 1 coors, 2 pn1_w, 3-6 pn1_bn(g,b,m,v), 7 view.pn_w, 8-11 view.pn_bn,
// 12 res1.w1, 13-16 res1.bn1, 17 res1.w2, 18-21 res1.bn2,
// 22 res2.w1, 23-26 res2.bn1, 27 res2.w2, 28-31 res2.bn2, 32 res2.wd, 33-36 res2.bnd,
// 37 res3.w1, 38-41 res3.bn1, 42 res3.w2, 43-46 res3.bn2, 47 res3.wd, 48-51 res3.bnd,
// 52 de2_w, 53 de3_w, 54 conv_w, 55 conv_b, 56 pn2_w, 57-60 pn2_bn, 61 pn3_w, 62-65 pn3_bn

extern "C" void kernel_launch(void* const* d_in, const int* in_sizes, int n_in,
                              void* d_out, int out_size, void* d_ws, size_t ws_size,
                              hipStream_t stream) {
  (void)in_sizes; (void)n_in; (void)ws_size;
  const float* feat  = (const float*)d_in[0];
  const int*   coors = (const int*)d_in[1];
#define IN_F(i) ((const float*)d_in[(i)])

  const int HP = BB * 200 * 176;  // half-res pixels
  const int QP = BB * 100 * 88;   // quarter-res pixels

  char* ws = (char*)d_ws;
  size_t off = 0;
  auto take = [&](size_t bytes) -> void* {
    void* p = ws + off;
    off = (off + bytes + 255) & ~(size_t)255;
    return p;
  };
  // --- swizzled bf16 weights ---
  bf16* wz_pn1  = (bf16*)take((size_t)32 * 128 * 2);
  bf16* wz_pn   = (bf16*)take((size_t)128 * 128 * 2);
  bf16* wz_pn2  = (bf16*)take((size_t)128 * 128 * 2);
  bf16* wz_pn3  = (bf16*)take((size_t)256 * 128 * 2);
  bf16* wz_r1w1 = (bf16*)take((size_t)9 * 16384 * 2);
  bf16* wz_r1w2 = (bf16*)take((size_t)9 * 16384 * 2);
  bf16* wz_r2w1 = (bf16*)take((size_t)9 * 16384 * 2);
  bf16* wz_r2w2 = (bf16*)take((size_t)9 * 16384 * 2);
  bf16* wz_r3w1 = (bf16*)take((size_t)9 * 16384 * 2);
  bf16* wz_r3w2 = (bf16*)take((size_t)9 * 16384 * 2);
  bf16* wz_r2wd = (bf16*)take((size_t)16384 * 2);
  bf16* wz_r3wd = (bf16*)take((size_t)16384 * 2);
  bf16* wz_de2  = (bf16*)take((size_t)4 * 16384 * 2);
  bf16* wz_de3  = (bf16*)take((size_t)16 * 16384 * 2);
  bf16* wz_fc   = (bf16*)take((size_t)27 * 16384 * 2);
  // --- activations / accumulators ---
  float* statbuf   = (float*)take((size_t)10 * NSEG * 4);
  float* cnt  = statbuf;
  float* sums = statbuf + NSEG;
  float* canvas_f  = (float*)take((size_t)NSEG * 128 * 4);
  bf16*  h_bf      = (bf16*)take((size_t)NPTS * 128 * 2);
  bf16*  canvas_bf = (bf16*)take((size_t)NSEG * 128 * 2);
  bf16*  t_full    = (bf16*)take((size_t)NSEG * 128 * 2);
  bf16*  out1      = (bf16*)take((size_t)NSEG * 128 * 2);
  bf16*  t_half    = (bf16*)take((size_t)HP * 128 * 2);
  bf16*  out2      = (bf16*)take((size_t)HP * 128 * 2);
  bf16*  idn2      = (bf16*)take((size_t)HP * 128 * 2);
  bf16*  t_q       = (bf16*)take((size_t)QP * 128 * 2);
  bf16*  out3      = (bf16*)take((size_t)QP * 128 * 2);
  bf16*  idn3      = (bf16*)take((size_t)QP * 128 * 2);
  bf16*  up2       = (bf16*)take((size_t)NSEG * 128 * 2);
  bf16*  up3       = (bf16*)take((size_t)NSEG * 128 * 2);
  bf16*  o_bf      = (bf16*)take((size_t)NSEG * 128 * 2);
  bf16*  h2_bf     = (bf16*)take((size_t)NPTS * 128 * 2);
  bf16*  pview_bf  = (bf16*)take((size_t)NPTS * 128 * 2);

  // --- weight prep ---
  auto prep = [&](const float* src, bf16* dst, int mode, int total, int xdim) {
    k_prep<<<(total + 255) / 256, 256, 0, stream>>>(src, dst, mode, total, xdim);
  };
  prep(IN_F(2),  wz_pn1, 0, 32 * 128, 20);
  prep(IN_F(7),  wz_pn,  0, 128 * 128, 128);
  prep(IN_F(56), wz_pn2, 0, 128 * 128, 128);
  prep(IN_F(61), wz_pn3, 0, 256 * 128, 256);
  prep(IN_F(12), wz_r1w1, 1, 9 * 16384, 0);
  prep(IN_F(17), wz_r1w2, 1, 9 * 16384, 0);
  prep(IN_F(22), wz_r2w1, 1, 9 * 16384, 0);
  prep(IN_F(27), wz_r2w2, 1, 9 * 16384, 0);
  prep(IN_F(37), wz_r3w1, 1, 9 * 16384, 0);
  prep(IN_F(42), wz_r3w2, 1, 9 * 16384, 0);
  prep(IN_F(32), wz_r2wd, 2, 16384, 0);
  prep(IN_F(47), wz_r3wd, 2, 16384, 0);
  prep(IN_F(52), wz_de2, 3, 4 * 16384, 2);
  prep(IN_F(53), wz_de3, 3, 16 * 16384, 4);
  prep(IN_F(54), wz_fc,  4, 27 * 16384, 0);

  // --- zero atomic accumulators ---
  hipMemsetAsync(statbuf, 0, (size_t)10 * NSEG * 4, stream);
  hipMemsetAsync(canvas_f, 0, (size_t)NSEG * 128 * 4, stream);
  hipMemsetAsync(d_out, 0, (size_t)out_size * sizeof(float), stream);

  // 1. segment stats
  k_scatter<<<(NPTS + 255) / 256, 256, 0, stream>>>(feat, coors, cnt, sums);

  // 2. pn1: stats -> h
  k_pn1<<<NPTS / 128, 256, 0, stream>>>(feat, coors, cnt, sums, wz_pn1, IN_F(3), IN_F(4),
                                        IN_F(5), IN_F(6), h_bf);

  // 3. pv = relu(bn(h @ pn_w)); pillar max into canvas_f
  k_gemm128<<<NPTS / 128, 256, 0, stream>>>(h_bf, wz_pn, IN_F(8), IN_F(9), IN_F(10), IN_F(11),
                                            1, nullptr, canvas_f, coors);
  k_cvt<<<(NSEG * 128) / 2048, 256, 0, stream>>>(canvas_f, canvas_bf);

  // 4. res1
  k_conv3<1, 32><<<BB * GNY * 11, 256, 0, stream>>>(canvas_bf, GNY, GNX, t_full, GNY, GNX,
      wz_r1w1, IN_F(13), IN_F(14), IN_F(15), IN_F(16), nullptr, 1);
  k_conv3<1, 32><<<BB * GNY * 11, 256, 0, stream>>>(t_full, GNY, GNX, out1, GNY, GNX,
      wz_r1w2, IN_F(18), IN_F(19), IN_F(20), IN_F(21), canvas_bf, 1);

  // 5. res2
  k_conv3<2, 16><<<BB * 200 * 11, 256, 0, stream>>>(canvas_bf, GNY, GNX, t_half, 200, 176,
      wz_r2w1, IN_F(23), IN_F(24), IN_F(25), IN_F(26), nullptr, 1);
  k_wd<<<BB * 200 * 6, 256, 0, stream>>>(canvas_bf, GNY, GNX, idn2, 200, 176,
      wz_r2wd, IN_F(33), IN_F(34), IN_F(35), IN_F(36));
  k_conv3<1, 32><<<BB * 200 * 6, 256, 0, stream>>>(t_half, 200, 176, out2, 200, 176,
      wz_r2w2, IN_F(28), IN_F(29), IN_F(30), IN_F(31), idn2, 1);

  // 6. res3
  k_conv3<2, 16><<<BB * 100 * 6, 256, 0, stream>>>(out2, 200, 176, t_q, 100, 88,
      wz_r3w1, IN_F(38), IN_F(39), IN_F(40), IN_F(41), nullptr, 1);
  k_wd<<<BB * 100 * 3, 256, 0, stream>>>(out2, 200, 176, idn3, 100, 88,
      wz_r3wd, IN_F(48), IN_F(49), IN_F(50), IN_F(51));
  k_conv3<1, 32><<<BB * 100 * 3, 256, 0, stream>>>(t_q, 100, 88, out3, 100, 88,
      wz_r3w2, IN_F(43), IN_F(44), IN_F(45), IN_F(46), idn3, 1);

  // 7. deconvs
  k_deconv<2><<<BB * 200 * 6, 256, 0, stream>>>(out2, 200, 176, up2, GNY, GNX, wz_de2);
  k_deconv<4><<<BB * 100 * 3, 256, 0, stream>>>(out3, 100, 88, up3, GNY, GNX, wz_de3);

  // 8. fusion conv
  k_fconv<<<BB * GNY * 11, 256, 0, stream>>>(out1, up2, up3, wz_fc, IN_F(55), o_bf);

  // 9. h2 = relu(bn(h @ pn2_w))
  k_gemm128<<<NPTS / 128, 256, 0, stream>>>(h_bf, wz_pn2, IN_F(57), IN_F(58), IN_F(59),
                                            IN_F(60), 0, h2_bf, nullptr, nullptr);

  // 10. bilinear gather
  k_gather<<<(NPTS * 16) / 256, 256, 0, stream>>>(feat, coors, o_bf, pview_bf);

  // 11. fm -> pillar max -> d_out
  k_pn3<<<NPTS / 128, 256, 0, stream>>>(pview_bf, h2_bf, wz_pn3, IN_F(62), IN_F(63),
                                        IN_F(64), IN_F(65), coors, (float*)d_out);
#undef IN_F
}
